// ODconv_24902220382412
// MI455X (gfx1250) — matrix operand order repaired and checked by reading, not compiled, not run on hardware
//
#include <hip/hip_runtime.h>
#include <hip/hip_bf16.h>

typedef _Float16 half_t;
typedef __attribute__((ext_vector_type(16))) _Float16 v16h;
typedef __attribute__((ext_vector_type(8)))  _Float16 v8h;
typedef __attribute__((ext_vector_type(2)))  __fp16   v2fp16;   // cvt_pkrtz return type
typedef __attribute__((ext_vector_type(8)))  float    v8f;

#define NB_DIM 12
#define RDIM   144          // 12*12 flattened (n,c)
#define KPAD   160          // 144 padded to 5 chunks of 32
#define HDIM   64
#define LDIM   64
#define XSZ    (RDIM * LDIM)   // 9216 elems per batch

#if __has_builtin(__builtin_amdgcn_wmma_f16_16x16x32_f16)
#define HAVE_WMMA_F16_OUT 1
#else
#define HAVE_WMMA_F16_OUT 0
#endif

union frag16 { v16h v; unsigned int u[8]; half_t h[16]; };

// One v_cvt_pk_rtz_f16_f32: packs two f32 into a dword of two f16 (RTZ)
__device__ __forceinline__ unsigned int pack2(float a, float b) {
    union { v2fp16 h; unsigned int u; } p;
    p.h = __builtin_amdgcn_cvt_pkrtz(a, b);
    return p.u;
}

// A-matrix fragment 16x32 (MxK), f16, row-major source with stride ld (halves).
// src points at element (row0, col0). Contiguous k order used here (not the ISA table order):
//   lanes 0-15: M=lane, K in {0..15}; lanes 16-31: M=lane-16, K in {16..31}
__device__ __forceinline__ v16h load_frag_a(const half_t* src, int ld, int lane) {
    frag16 f;
    const int lo = lane & 15, hi = (lane >> 4) & 1;
    const half_t* rp = src + lo * ld;
#pragma unroll
    for (int v = 0; v < 8; ++v) {
        const int kbase = 2 * v + hi * 16;
        f.u[v] = *(const unsigned int*)(rp + kbase);   // contiguous 16B runs -> b128 loads
    }
    return f.v;
}

// B-matrix fragment 32x16 (KxN), f16, from a K-CONTIGUOUS ([N][K]) source with
// row stride ld (halves). src points at element (n0, k0). Lane's 16 halves are
// K = hi*16 + [0..15] at fixed N = n0+lo: one contiguous 32B aligned run.
__device__ __forceinline__ v16h load_frag_b_kc(const half_t* src, int ld, int lane) {
    const int lo = lane & 15, hi = (lane >> 4) & 1;
    return *(const v16h*)(src + lo * ld + hi * 16);
}

__device__ __forceinline__ v8f wmma_f32(v16h a, v16h b, v8f c) {
    return __builtin_amdgcn_wmma_f32_16x16x32_f16(false, a, false, b, (short)0, c, false, false);
}

// ---------------------------------------------------------------------------
// Prep: compute M = res_b (12x12), build T (144x160 f16, K-padded, row-major)
// and W2h (64x64 f16, SAME layout as W2: [h][l] -> K(l)-contiguous B operand)
// ---------------------------------------------------------------------------
__global__ __launch_bounds__(128) void odconv_prep(
        const float* __restrict__ G, const float* __restrict__ G_add,
        const float* __restrict__ W2, const float* __restrict__ attn_w,
        const float* __restrict__ attn_b,
        half_t* __restrict__ Th, half_t* __restrict__ W2h) {
    __shared__ float M[RDIM];
    const int tid = threadIdx.x;

    // attG = concat(G_add(2,3,12,12), G(3,12,12)) ; btmp[k] = attG[k,1]
    const float* bt0 = G_add + 144;            // [0,1,:,:]
    const float* bt1 = G_add + 3 * 144 + 144;  // [1,1,:,:]
    const float* bt2 = G + 144;                // G[1,:,:]

    // Redundant per-thread scalar attention (tiny)
    float s0 = 0.f, s1 = 0.f, s2 = 0.f;
    for (int i = 0; i < 144; ++i) {
        const float w = attn_w[i];
        s0 += bt0[i] * w; s1 += bt1[i] * w; s2 += bt2[i] * w;
    }
    const float ab = attn_b[0];
    s0 += ab; s1 += ab; s2 += ab;
    const float mx = fmaxf(s0, fmaxf(s1, s2));
    const float e0 = expf(s0 - mx), e1 = expf(s1 - mx), e2 = expf(s2 - mx);
    const float inv = 1.0f / (e0 + e1 + e2);
    const float w0 = e0 * inv, w1 = e1 * inv, w2 = e2 * inv;

    for (int e = tid; e < 144; e += blockDim.x)
        M[e] = w0 * bt0[e] + w1 * bt1[e] + w2 * bt2[e];
    __syncthreads();

    // T[(m,d),(n,c)] = delta + M[n,m]*M[c,d], K-padded with zeros to 160
    for (int idx = tid; idx < RDIM * KPAD; idx += blockDim.x) {
        const int r = idx / KPAD, ci = idx % KPAD;
        float t = 0.f;
        if (ci < RDIM) {
            const int m = r / NB_DIM, d = r % NB_DIM;
            const int n = ci / NB_DIM, c = ci % NB_DIM;
            t = M[n * NB_DIM + m] * M[c * NB_DIM + d] + ((m == n && d == c) ? 1.0f : 0.0f);
        }
        Th[idx] = (half_t)t;
    }
    // W2h = (f16)W2, unchanged layout: row h, K(l) contiguous
    for (int idx = tid; idx < HDIM * LDIM; idx += blockDim.x)
        W2h[idx] = (half_t)W2[idx];
}

// ---------------------------------------------------------------------------
// Main: per batch b: A = Xh[b] @ W2^T (144x64), Y = T @ A + b2 (144x64, f32)
// 128 threads = 4 wave32; waves split the 9 row tiles.
// Intermediate A kept TRANSPOSED in LDS: Aht[h][k], k in [0,160).
// ---------------------------------------------------------------------------
__global__ __launch_bounds__(128) void odconv_main(
        const float* __restrict__ X, const half_t* __restrict__ Th,
        const half_t* __restrict__ W2h, const float* __restrict__ b2,
        float* __restrict__ out) {
    __shared__ half_t Xs[RDIM * LDIM];     // 18 KB : X[b] in f16, row-major [r][l]
    __shared__ half_t W2s[HDIM * LDIM];    //  8 KB : (f16)W2, [h][l]
    __shared__ half_t Aht[HDIM * KPAD];    // 20 KB : A^T, [h][k], K-padded

    const int b   = blockIdx.x;
    const int tid = threadIdx.x;
    const int wave = tid >> 5, lane = tid & 31;
    const int lo = lane & 15, hi = (lane >> 4) & 1;
    const float* xb = X + (size_t)b * XSZ;

    // ---- stage X[b] -> f16 LDS (coalesced float4 in, b64 out) ----
#pragma unroll
    for (int i = 0; i < XSZ / 4 / 128; ++i) {          // 18 iters
        const int idx = tid + i * 128;                 // float4 index
        const float4 xv = ((const float4*)xb)[idx];
        uint2 p; p.x = pack2(xv.x, xv.y); p.y = pack2(xv.z, xv.w);
        ((uint2*)Xs)[idx] = p;
    }
    // ---- stage W2 (f16) ----
#pragma unroll
    for (int i = 0; i < (HDIM * LDIM / 2) / 128; ++i)  // 16 iters of u32
        ((unsigned int*)W2s)[tid + i * 128] = ((const unsigned int*)W2h)[tid + i * 128];
    // ---- zero pad columns k in [144,160) of Aht ----
#pragma unroll
    for (int i = 0; i < (HDIM * 16 / 2) / 128; ++i) {  // 4 iters of u32
        const int j = tid + i * 128;                   // 512 u32 slots
        const int n = j >> 3, kk = (j & 7) * 2;
        *(unsigned int*)(Aht + n * KPAD + RDIM + kk) = 0u;
    }
    __syncthreads();

    // ---- GEMM1: A^T = (Xs(144x64) @ W2s^T)^T, f16 into Aht ----
    v16h bf1[2][4];
#pragma unroll
    for (int k = 0; k < 2; ++k)
#pragma unroll
        for (int h = 0; h < 4; ++h)
            bf1[k][h] = load_frag_b_kc(W2s + (h * 16) * LDIM + k * 32, LDIM, lane);

    for (int t = wave; t < 9; t += 4) {
        const v16h a0 = load_frag_a(Xs + t * 16 * LDIM + 0,  LDIM, lane);
        const v16h a1 = load_frag_a(Xs + t * 16 * LDIM + 32, LDIM, lane);
#pragma unroll
        for (int h = 0; h < 4; ++h) {
#if HAVE_WMMA_F16_OUT
            // f16 C/D layout: lane(hi)'s 8 halves = rows t*16+hi*8+[0..7] at
            // col h*16+lo -> exactly the 8 contiguous Aht halves, pre-packed.
            v8h ch = {};
            ch = __builtin_amdgcn_wmma_f16_16x16x32_f16(false, a0, false, bf1[0][h],
                                                        (short)0, ch, false, false);
            ch = __builtin_amdgcn_wmma_f16_16x16x32_f16(false, a1, false, bf1[1][h],
                                                        (short)0, ch, false, false);
            *(v8h*)(Aht + (h * 16 + lo) * KPAD + t * 16 + hi * 8) = ch;
#else
            v8f c = {};
            c = wmma_f32(a0, bf1[0][h], c);
            c = wmma_f32(a1, bf1[1][h], c);
            uint4 p;
            p.x = pack2(c[0], c[1]); p.y = pack2(c[2], c[3]);
            p.z = pack2(c[4], c[5]); p.w = pack2(c[6], c[7]);
            *(uint4*)(Aht + (h * 16 + lo) * KPAD + t * 16 + hi * 8) = p;
#endif
        }
    }
    __syncthreads();

    // ---- GEMM2: out[b] = T(144x160) @ A(160x64) + b2 ----
    float bias[4];
#pragma unroll
    for (int h = 0; h < 4; ++h) bias[h] = b2[h * 16 + lo];

    for (int t = wave; t < 9; t += 4) {
        v16h ta[5];
#pragma unroll
        for (int kc = 0; kc < 5; ++kc)
            ta[kc] = load_frag_a(Th + t * 16 * KPAD + kc * 32, KPAD, lane);
#pragma unroll
        for (int h = 0; h < 4; ++h) {
            v8f c = {};
#pragma unroll
            for (int kc = 0; kc < 5; ++kc) {
                const v16h bb = load_frag_b_kc(Aht + (h * 16) * KPAD + kc * 32, KPAD, lane);
                c = wmma_f32(ta[kc], bb, c);
            }
            float* ob = out + (size_t)b * XSZ + (t * 16) * HDIM + h * 16;
#pragma unroll
            for (int v = 0; v < 8; ++v)
                ob[(v + hi * 8) * HDIM + lo] = c[v] + bias[h];
        }
    }
}

extern "C" void kernel_launch(void* const* d_in, const int* in_sizes, int n_in,
                              void* d_out, int out_size, void* d_ws, size_t ws_size,
                              hipStream_t stream) {
    const float* X      = (const float*)d_in[0];
    const float* G      = (const float*)d_in[1];
    const float* G_add  = (const float*)d_in[2];
    const float* W2     = (const float*)d_in[3];
    const float* b2     = (const float*)d_in[4];
    const float* attn_w = (const float*)d_in[5];
    const float* attn_b = (const float*)d_in[6];

    half_t* Th  = (half_t*)d_ws;           // 144*160 halves
    half_t* W2h = Th + RDIM * KPAD;        // 64*64 halves

    const int nB = in_sizes[0] / XSZ;      // 8192

    odconv_prep<<<1, 128, 0, stream>>>(G, G_add, W2, attn_w, attn_b, Th, W2h);
    odconv_main<<<nB, 128, 0, stream>>>(X, Th, W2h, b2, (float*)d_out);
}
